// TransformerBlock_59261958750607
// MI455X (gfx1250) — matrix operand order repaired and checked by reading, not compiled, not run on hardware
//
#include <hip/hip_runtime.h>
#include <hip/hip_bf16.h>
#include <math.h>

// ---------------------------------------------------------------------------
// Problem constants (from reference): B=2, S=2048, D=768, H=12, HD=64, M=3072
// ---------------------------------------------------------------------------
#define BB   2
#define SS   2048
#define DD   768
#define HH   12
#define HD   64
#define MM   3072
#define ROWS (BB * SS)          // 4096 token rows
#define QKV_STRIDE ((size_t)BB * HH * SS * HD)   // elements per q/k/v buffer

typedef __attribute__((ext_vector_type(16))) _Float16 v16h;
typedef __attribute__((ext_vector_type(8)))  _Float16 v8h;
typedef __attribute__((ext_vector_type(8)))  float    v8f;

// ---------------------------------------------------------------------------
// A-fragment loader: 16-bit A matrix 16x32 (contiguous k order used by this code, matching load_b16).
// lane M = lane&15 ; kbase = (lane<16)?0:16
// elements 0..7  -> K = kbase+0 .. kbase+7      (contiguous 16B)
// elements 8..15 -> K = kbase+8 .. kbase+15     (contiguous 16B)
// ---------------------------------------------------------------------------
__device__ __forceinline__ v16h load_a16(const _Float16* rowp, int kbase) {
    union U { v16h v; v8h h[2]; } u;
    u.h[0] = *(const v8h*)(rowp + kbase);
    u.h[1] = *(const v8h*)(rowp + kbase + 8);
    return u.v;
}

// B-fragment: lane N = lane&15 ; kbase2 = (lane<16)?0:16 ; 16 contiguous K
// (requires B staged transposed: [N][K] halves, 32B-aligned rows)
__device__ __forceinline__ v16h load_b16(const _Float16* np, int kbase2) {
    return *(const v16h*)(np + kbase2);
}

// ---------------------------------------------------------------------------
// Weight convert + transpose: out[N][K] (f16) <- in[K][N] (f32).
// LDS-tiled 32x32 so both the global read and the global write are coalesced.
// ---------------------------------------------------------------------------
__global__ __launch_bounds__(256) void cvt_transpose_kernel(
    const float* __restrict__ in, _Float16* __restrict__ out, int K, int N) {
    __shared__ float tile[32][33];
    const int bn = blockIdx.x * 32;     // column block (N)
    const int bk = blockIdx.y * 32;     // row block (K)
    const int tx = threadIdx.x & 31;
    const int ty = threadIdx.x >> 5;    // 0..7
    #pragma unroll
    for (int i = 0; i < 32; i += 8)
        tile[ty + i][tx] = in[(size_t)(bk + ty + i) * N + bn + tx];
    __syncthreads();
    #pragma unroll
    for (int i = 0; i < 32; i += 8)
        out[(size_t)(bn + ty + i) * K + bk + tx] = (_Float16)tile[tx][ty + i];
}

// ---------------------------------------------------------------------------
// LayerNorm: one block (256 threads = 8 waves) per token row. f32 in, f16 out.
// ---------------------------------------------------------------------------
__global__ __launch_bounds__(256) void layernorm_f16_kernel(
    const float* __restrict__ x, const float* __restrict__ g,
    const float* __restrict__ b, _Float16* __restrict__ out) {
    const int row = blockIdx.x;
    const float* xr = x + (size_t)row * DD;
    float s = 0.f, s2 = 0.f;
    for (int i = threadIdx.x; i < DD; i += 256) {
        float v = xr[i];
        s += v; s2 += v * v;
    }
    for (int off = 16; off > 0; off >>= 1) {
        s  += __shfl_down(s,  off, 32);
        s2 += __shfl_down(s2, off, 32);
    }
    __shared__ float red[2][8];
    const int lane = threadIdx.x & 31, w = threadIdx.x >> 5;
    if (lane == 0) { red[0][w] = s; red[1][w] = s2; }
    __syncthreads();
    if (w == 0) {
        s  = (lane < 8) ? red[0][lane] : 0.f;
        s2 = (lane < 8) ? red[1][lane] : 0.f;
        for (int off = 4; off > 0; off >>= 1) {
            s  += __shfl_down(s,  off, 32);
            s2 += __shfl_down(s2, off, 32);
        }
        if (lane == 0) { red[0][0] = s; red[1][0] = s2; }
    }
    __syncthreads();
    const float mu  = red[0][0] * (1.0f / DD);
    const float var = red[1][0] * (1.0f / DD) - mu * mu;
    const float inv = rsqrtf(var + 1e-5f);
    for (int i = threadIdx.x; i < DD; i += 256) {
        float v = (xr[i] - mu) * inv * g[i] + b[i];
        out[(size_t)row * DD + i] = (_Float16)v;
    }
}

// ---------------------------------------------------------------------------
// WMMA GEMM: C[M x N] = A[M x K](f16) @ W(f16, pre-transposed [N][K]) + bias.
//   mode 0: QKV scatter -> q,k: [B,H,S,HD]; v: [B,H,HD,S] (f16)
//   mode 1: f32 out = acc + bias + res            (residual add)
//   mode 2: f16 out = GELU_exact(acc + bias)
// 128 thr = 4 waves; block tile 128x128, BK=32; wave tile 64x64 = 4x4 WMMA
// (16 WMMAs per 8 fragment loads). Double-buffered LDS, peeled pipeline:
// unconditional next-tile loads, one barrier per K-step, no zero-init waste.
// ---------------------------------------------------------------------------
#define Bb 128
#define BN 128
#define BK 32
#define GT 128

__global__ __launch_bounds__(GT) void gemm_wmma_kernel(
    const _Float16* __restrict__ A, const _Float16* __restrict__ Wt,
    const float* __restrict__ bias, const float* __restrict__ res,
    void* __restrict__ out, _Float16* __restrict__ qkv_out,
    int Ndim, int Kdim, int mode) {

    __shared__ __align__(64) _Float16 As[2][Bb * BK];  // [m][k]
    __shared__ __align__(64) _Float16 Bs[2][BN * BK];  // [n][k]

    const int tid  = threadIdx.x;
    const int lane = tid & 31;
    const int w    = tid >> 5;          // 0..3
    const int wm   = (w & 1) * 64;      // 64-row band
    const int wn   = (w >> 1) * 64;     // 64-col band
    const int rowBase = blockIdx.y * Bb;
    const int colBase = blockIdx.x * BN;

    const int kba = (lane < 16) ? 0 : 16;
    const int kbb = (lane < 16) ? 0 : 16;

    // staging: thread covers rows r, r+32, r+64, r+96; 8 halves at col c
    const int r = tid >> 2;             // 0..31
    const int c = (tid & 3) * 8;        // halves
    const _Float16* gA = A  + (size_t)(rowBase + r) * Kdim + c;
    const _Float16* gB = Wt + (size_t)(colBase + r) * Kdim + c;
    const size_t strideR = (size_t)32 * Kdim;

    // ---- prologue: stage K-step 0 into buffer 0 ----
    #pragma unroll
    for (int q = 0; q < 4; ++q) {
        *(v8h*)(As[0] + (r + q * 32) * BK + c) = *(const v8h*)(gA + q * strideR);
        *(v8h*)(Bs[0] + (r + q * 32) * BK + c) = *(const v8h*)(gB + q * strideR);
    }
    __syncthreads();

    v8f acc[4][4] = {};
    const int nk = Kdim / BK;

    auto compute = [&](const _Float16* as, const _Float16* bs) {
        v16h a[4];
        #pragma unroll
        for (int i = 0; i < 4; ++i)
            a[i] = load_a16(as + (size_t)(wm + i * 16 + (lane & 15)) * BK, kba);
        #pragma unroll
        for (int j = 0; j < 4; ++j) {
            v16h b = load_b16(bs + (size_t)(wn + j * 16 + (lane & 15)) * BK, kbb);
            #pragma unroll
            for (int i = 0; i < 4; ++i)
                acc[i][j] = __builtin_amdgcn_wmma_f32_16x16x32_f16(
                    false, a[i], false, b, (short)0, acc[i][j], false, false);
        }
    };

    // ---- main pipeline: loads are unconditional (last step peeled) ----
    for (int t = 0; t < nk - 1; ++t) {
        const size_t ko = (size_t)(t + 1) * BK;
        v8h ra[4], rb[4];
        #pragma unroll
        for (int q = 0; q < 4; ++q) {
            ra[q] = *(const v8h*)(gA + q * strideR + ko);
            rb[q] = *(const v8h*)(gB + q * strideR + ko);
        }
        if (t + 2 < nk) {
            __builtin_prefetch(gA + ko + BK, 0, 3);
            __builtin_prefetch(gB + ko + BK, 0, 3);
        }

        compute(As[t & 1], Bs[t & 1]);

        _Float16* an = As[(t + 1) & 1];
        _Float16* bn = Bs[(t + 1) & 1];
        #pragma unroll
        for (int q = 0; q < 4; ++q) {
            *(v8h*)(an + (r + q * 32) * BK + c) = ra[q];
            *(v8h*)(bn + (r + q * 32) * BK + c) = rb[q];
        }
        __syncthreads();
    }
    compute(As[(nk - 1) & 1], Bs[(nk - 1) & 1]);

    // ---- epilogue (C layout: VGPR rr -> M = rr + (lane>=16 ? 8:0), N = lane&15) ----
    const int rofs = (lane < 16) ? 0 : 8;
    #pragma unroll
    for (int i = 0; i < 4; ++i) {
        #pragma unroll
        for (int j = 0; j < 4; ++j) {
            const int tileM = rowBase + wm + i * 16;
            const int col   = colBase + wn + j * 16 + (lane & 15);
            const float bv  = bias[col];
            #pragma unroll
            for (int rr = 0; rr < 8; ++rr) {
                const int row = tileM + rr + rofs;
                float v = acc[i][j][rr] + bv;
                if (mode == 0) {
                    // qkv row-vector layout is [H][3][HD]
                    const int head  = col / (3 * HD);
                    const int which = (col / HD) % 3;
                    const int hd    = col % HD;
                    const int bb    = row >> 11;          // /2048
                    const int ss    = row & 2047;
                    size_t idx;
                    if (which == 2)  // V stored head-transposed [B,H,HD,S]
                        idx = 2 * QKV_STRIDE + ((size_t)(bb * HH + head) * HD + hd) * SS + ss;
                    else
                        idx = (size_t)which * QKV_STRIDE +
                              ((size_t)(bb * HH + head) * SS + ss) * HD + hd;
                    qkv_out[idx] = (_Float16)v;
                } else if (mode == 1) {
                    const size_t idx = (size_t)row * Ndim + col;
                    ((float*)out)[idx] = v + res[idx];
                } else {
                    const float ge = 0.5f * v * (1.0f + erff(v * 0.70710678118654752f));
                    ((_Float16*)out)[(size_t)row * Ndim + col] = (_Float16)ge;
                }
            }
        }
    }
}

// ---------------------------------------------------------------------------
// Flash attention (online softmax), one block per (head, 128-query tile).
// 256 threads = 8 waves; each wave owns a 16-query band.
// V arrives pre-transposed ([B,H,HD,S]) so all staging is vectorized copies.
// LDS (dynamic, ~177 KB; CDNA5 has 320 KB/WGP).
// ---------------------------------------------------------------------------
#define TQ 128
#define TK 128

extern __shared__ __align__(128) char fa_smem[];

__global__ __launch_bounds__(256) void flash_attn_kernel(
    const _Float16* __restrict__ qb, const _Float16* __restrict__ kb,
    const _Float16* __restrict__ vbt, _Float16* __restrict__ ob) {

    _Float16* Qs = (_Float16*)fa_smem;            // [128][64]
    _Float16* Ks = Qs + TQ * HD;                  // [128][64]
    _Float16* Vt = Ks + TK * HD;                  // [64][128] (hd-major)
    _Float16* Ps = Vt + HD * TK;                  // [128][128]
    float*    Sb = (float*)(Ps + TQ * TK);        // [128][128]
    float*    Ob = Sb + TQ * TK;                  // [128][64]
    float*    mrow = Ob + TQ * HD;                // [128]
    float*    lrow = mrow + TQ;                   // [128]

    const int tid  = threadIdx.x;
    const int lane = tid & 31;
    const int w    = tid >> 5;                    // wave 0..7
    const int bh   = blockIdx.y;                  // 0..23 (b*H + h)
    const int qt   = blockIdx.x;                  // 0..15

    const size_t headBase = (size_t)bh * SS * HD; // same count for q/k and v^T
    const int kba = (lane < 16) ? 0 : 16;
    const int kbb = (lane < 16) ? 0 : 16;

    // ---- load Q tile, init O / stats ----
    {
        const v8h* src = (const v8h*)(qb + headBase + (size_t)qt * TQ * HD);
        for (int i = tid; i < TQ * HD / 8; i += 256) ((v8h*)Qs)[i] = src[i];
        for (int i = tid; i < TQ * HD; i += 256) Ob[i] = 0.f;
        if (tid < TQ) { mrow[tid] = -1e30f; lrow[tid] = 0.f; }
    }
    __syncthreads();

    for (int kt = 0; kt < SS / TK; ++kt) {
        // ---- stage K tile [key][hd] and V^T tile [hd][key] (all vectorized) ----
        {
            const v8h* ksrc = (const v8h*)(kb + headBase + (size_t)kt * TK * HD);
            for (int i = tid; i < TK * HD / 8; i += 256) ((v8h*)Ks)[i] = ksrc[i];
            // vbt rows are whole-S key streams: contiguous copy per hd row
            for (int i = tid; i < HD * TK / 8; i += 256) {
                const int hd = i >> 4;            // row (TK/8 = 16 chunks)
                const int ch = i & 15;
                *(v8h*)(Vt + (size_t)hd * TK + ch * 8) =
                    *(const v8h*)(vbt + headBase + (size_t)hd * SS + kt * TK + ch * 8);
            }
        }
        __syncthreads();

        // ---- S = (Q @ K^T) * 1/sqrt(HD) : wave w -> rows [16w,16w+16) x 128 ----
        {
            v8f sacc[8] = {};
            const _Float16* arow = Qs + (size_t)(w * 16 + (lane & 15)) * HD;
            #pragma unroll
            for (int kc = 0; kc < HD / 32; ++kc) {
                v16h a = load_a16(arow + kc * 32, kba);
                #pragma unroll
                for (int nt = 0; nt < 8; ++nt) {
                    const int key = nt * 16 + (lane & 15);
                    v16h bfr = load_b16(Ks + (size_t)key * HD + kc * 32, kbb);
                    sacc[nt] = __builtin_amdgcn_wmma_f32_16x16x32_f16(
                        false, a, false, bfr, (short)0, sacc[nt], false, false);
                }
            }
            const int rr0 = w * 16 + ((lane < 16) ? 0 : 8);
            #pragma unroll
            for (int nt = 0; nt < 8; ++nt) {
                const int cc = nt * 16 + (lane & 15);
                #pragma unroll
                for (int rr = 0; rr < 8; ++rr)
                    Sb[(size_t)(rr0 + rr) * TK + cc] = sacc[nt][rr] * 0.125f;
            }
        }
        __syncthreads();

        // ---- online softmax: 2 threads per query row, shfl_xor combine ----
        {
            const int row  = tid >> 1;
            const int half = tid & 1;
            const float* srow = Sb + (size_t)row * TK + half * 64;
            const float mold = mrow[row];
            float mx = -1e30f;
            for (int cix = 0; cix < 64; ++cix) mx = fmaxf(mx, srow[cix]);
            mx = fmaxf(mx, __shfl_xor(mx, 1, 32));
            mx = fmaxf(mx, mold);
            float sum = 0.f;
            _Float16* prow = Ps + (size_t)row * TK + half * 64;
            for (int cix = 0; cix < 64; ++cix) {
                const float p = __expf(srow[cix] - mx);
                sum += p;
                prow[cix] = (_Float16)p;
            }
            sum += __shfl_xor(sum, 1, 32);
            const float corr = __expf(mold - mx);
            if (half == 0) {
                mrow[row] = mx;
                lrow[row] = lrow[row] * corr + sum;
            }
            float* orow = Ob + (size_t)row * HD + half * 32;
            for (int cix = 0; cix < 32; ++cix) orow[cix] *= corr;
        }
        __syncthreads();

        // ---- O += P @ V : wave w -> rows [16w,16w+16) x 64 ----
        {
            v8f oacc[4] = {};
            const _Float16* arow = Ps + (size_t)(w * 16 + (lane & 15)) * TK;
            #pragma unroll
            for (int kc = 0; kc < TK / 32; ++kc) {
                v16h a = load_a16(arow + kc * 32, kba);
                #pragma unroll
                for (int nt = 0; nt < 4; ++nt) {
                    const int hd = nt * 16 + (lane & 15);
                    v16h bfr = load_b16(Vt + (size_t)hd * TK + kc * 32, kbb);
                    oacc[nt] = __builtin_amdgcn_wmma_f32_16x16x32_f16(
                        false, a, false, bfr, (short)0, oacc[nt], false, false);
                }
            }
            const int rr0 = w * 16 + ((lane < 16) ? 0 : 8);
            #pragma unroll
            for (int nt = 0; nt < 4; ++nt) {
                const int cc = nt * 16 + (lane & 15);
                #pragma unroll
                for (int rr = 0; rr < 8; ++rr)
                    Ob[(size_t)(rr0 + rr) * HD + cc] += oacc[nt][rr];
            }
        }
        __syncthreads();
    }

    // ---- normalize & write o into [B,S,D] layout (f16), 2 threads per row ----
    {
        const int row  = tid >> 1;
        const int half = tid & 1;
        const float invl = 1.0f / lrow[row];
        const int grow = (bh / HH) * SS + qt * TQ + row;
        _Float16* dst = ob + (size_t)grow * DD + (bh % HH) * HD + half * 32;
        const float* orow = Ob + (size_t)row * HD + half * 32;
        for (int cix = 0; cix < 32; ++cix) dst[cix] = (_Float16)(orow[cix] * invl);
    }
}

// ---------------------------------------------------------------------------
// Host-side orchestration
// ---------------------------------------------------------------------------
extern "C" void kernel_launch(void* const* d_in, const int* in_sizes, int n_in,
                              void* d_out, int out_size, void* d_ws, size_t ws_size,
                              hipStream_t stream) {
    (void)in_sizes; (void)n_in; (void)out_size; (void)ws_size;

    const float* x     = (const float*)d_in[0];
    const float* qkv_w = (const float*)d_in[1];
    const float* qkv_b = (const float*)d_in[2];
    const float* out_w = (const float*)d_in[3];
    const float* out_b = (const float*)d_in[4];
    const float* ln1_g = (const float*)d_in[5];
    const float* ln1_b = (const float*)d_in[6];
    const float* ln2_g = (const float*)d_in[7];
    const float* ln2_b = (const float*)d_in[8];
    const float* w1    = (const float*)d_in[9];
    const float* b1    = (const float*)d_in[10];
    const float* w2    = (const float*)d_in[11];
    const float* b2    = (const float*)d_in[12];
    float* out = (float*)d_out;

    // --- workspace carve (256B aligned) ---
    char* ws = (char*)d_ws;
    size_t off = 0;
    auto carve = [&](size_t bytes) -> void* {
        void* p = ws + off;
        off = (off + bytes + 255) & ~(size_t)255;
        return p;
    };
    _Float16* qkv_wT = (_Float16*)carve((size_t)DD * 3 * DD * 2);  // [3D][D]
    _Float16* out_wT = (_Float16*)carve((size_t)DD * DD * 2);      // [D][D]
    _Float16* w1_T   = (_Float16*)carve((size_t)DD * MM * 2);      // [M][D]
    _Float16* w2_T   = (_Float16*)carve((size_t)MM * DD * 2);      // [D][M]
    _Float16* h16    = (_Float16*)carve((size_t)ROWS * DD * 2);
    _Float16* qkvbuf = (_Float16*)carve(3 * QKV_STRIDE * 2);       // q|k|v^T
    _Float16* obuf   = (_Float16*)carve((size_t)ROWS * DD * 2);
    float*    x1     = (float*)   carve((size_t)ROWS * DD * 4);
    _Float16* h2_16  = (_Float16*)carve((size_t)ROWS * DD * 2);
    _Float16* hid16  = (_Float16*)carve((size_t)ROWS * MM * 2);

    // --- 1) weight convert + transpose: Wt[N][K] f16 <- W[K][N] f32 ---
    auto cvtT = [&](const float* src, _Float16* dst, int K, int N) {
        cvt_transpose_kernel<<<dim3(N / 32, K / 32), 256, 0, stream>>>(src, dst, K, N);
    };
    cvtT(qkv_w, qkv_wT, DD, 3 * DD);
    cvtT(out_w, out_wT, DD, DD);
    cvtT(w1,    w1_T,   DD, MM);
    cvtT(w2,    w2_T,   MM, DD);

    // --- 2) h = LN1(x) (f16) ---
    layernorm_f16_kernel<<<ROWS, 256, 0, stream>>>(x, ln1_g, ln1_b, h16);

    // --- 3) qkv = h @ qkv_w + qkv_b  (scatter; V head-transposed) ---
    gemm_wmma_kernel<<<dim3(3 * DD / BN, ROWS / Bb), GT, 0, stream>>>(
        h16, qkv_wT, qkv_b, nullptr, nullptr, qkvbuf, 3 * DD, DD, 0);

    // --- 4) flash attention -> obuf [B,S,D] f16 ---
    const size_t fa_lds =
        (size_t)(TQ * HD * 3 + TQ * TK) * 2 +                  // f16 regions
        (size_t)(TQ * TK + TQ * HD + 2 * TQ) * 4;              // f32 regions
    hipFuncSetAttribute((const void*)flash_attn_kernel,
                        hipFuncAttributeMaxDynamicSharedMemorySize, (int)fa_lds);
    flash_attn_kernel<<<dim3(SS / TQ, BB * HH), 256, fa_lds, stream>>>(
        qkvbuf, qkvbuf + QKV_STRIDE, qkvbuf + 2 * QKV_STRIDE, obuf);

    // --- 5) x1 = x + obuf @ out_w + out_b (f32) ---
    gemm_wmma_kernel<<<dim3(DD / BN, ROWS / Bb), GT, 0, stream>>>(
        obuf, out_wT, out_b, x, x1, nullptr, DD, DD, 1);

    // --- 6) h2 = LN2(x1) (f16) ---
    layernorm_f16_kernel<<<ROWS, 256, 0, stream>>>(x1, ln2_g, ln2_b, h2_16);

    // --- 7) hid = GELU(h2 @ w1 + b1) (f16) ---
    gemm_wmma_kernel<<<dim3(MM / BN, ROWS / Bb), GT, 0, stream>>>(
        h2_16, w1_T, b1, nullptr, hid16, nullptr, MM, DD, 2);

    // --- 8) out = x1 + hid @ w2 + b2 (f32) ---
    gemm_wmma_kernel<<<dim3(DD / BN, ROWS / Bb), GT, 0, stream>>>(
        hid16, w2_T, b2, x1, out, nullptr, DD, MM, 1);
}
